// AttachmentPredictor_7095285973512
// MI455X (gfx1250) — matrix operand order repaired and checked by reading, not compiled, not run on hardware
//
#include <hip/hip_runtime.h>
#include <hip/hip_bf16.h>
#include <math.h>

// ---------------------------------------------------------------------------
// Problem constants (from reference): B=256, S=256, D=1024, P=512, SH=254
// ---------------------------------------------------------------------------
#define CB  256
#define CS  256
#define CD  1024
#define CP  512
#define CSH 254
#define EPSV 1e-7f

typedef __attribute__((ext_vector_type(16))) __bf16 v16bf;
typedef __attribute__((ext_vector_type(8)))  __bf16 v8bf;
typedef __attribute__((ext_vector_type(4)))  __bf16 v4bf;
typedef __attribute__((ext_vector_type(8)))  float  v8f;

union Frag16 { v16bf v; v8bf h[2]; };

// Fast tanh via hardware v_exp_f32: tanh(x) = 1 - 2/(exp(2x)+1)
// (exact at saturation: exp->0 gives -1, exp->inf gives +1)
__device__ __forceinline__ float fast_tanh(float v) {
    return 1.0f - 2.0f / (__expf(2.0f * v) + 1.0f);
}

// ---------------------------------------------------------------------------
// Kernel 1: fp32 [K][N] -> bf16 transposed [N][K]
// ---------------------------------------------------------------------------
__global__ void transpose_to_bf16(const float* __restrict__ in,
                                  __bf16* __restrict__ out,
                                  int K, int N) {
    int idx = blockIdx.x * blockDim.x + threadIdx.x;
    if (idx >= K * N) return;
    int n = idx / K;
    int k = idx - n * K;
    out[idx] = (__bf16)in[(size_t)k * N + n];
}

// ---------------------------------------------------------------------------
// Kernel 2: pc[b][p] = sum_d x[b,S-2,d]*Wp[d][p] + x[b,S-1,d]*Wc[d][p]
// grid (2, B/16), block 256. Each block: 256 p-columns x 16 batches.
// Weight columns read once per 16 batches -> 16x less L2 traffic.
// ---------------------------------------------------------------------------
__global__ __launch_bounds__(256)
void pc_kernel(const float* __restrict__ x,
               const float* __restrict__ Wp,
               const float* __restrict__ Wc,
               float* __restrict__ pc) {
    __shared__ float sPrep[16][128];
    __shared__ float sChild[16][128];
    const int p  = blockIdx.x * 256 + threadIdx.x;
    const int b0 = blockIdx.y * 16;

    float acc[16];
#pragma unroll
    for (int bb = 0; bb < 16; ++bb) acc[bb] = 0.0f;

    for (int dc = 0; dc < CD; dc += 128) {
        __syncthreads();
        // stage 16 batches x 128 d of prep & child rows (coalesced)
#pragma unroll
        for (int it = 0; it < 8; ++it) {
            int idx = threadIdx.x + it * 256;      // 0..2047
            int bb  = idx >> 7;
            int d   = idx & 127;
            sPrep[bb][d]  = x[((size_t)(b0 + bb) * CS + (CS - 2)) * CD + dc + d];
            sChild[bb][d] = x[((size_t)(b0 + bb) * CS + (CS - 1)) * CD + dc + d];
        }
        __syncthreads();
#pragma unroll 4
        for (int d = 0; d < 128; ++d) {
            float wp = Wp[(size_t)(dc + d) * CP + p];
            float wc = Wc[(size_t)(dc + d) * CP + p];
#pragma unroll
            for (int bb = 0; bb < 16; ++bb)
                acc[bb] += sPrep[bb][d] * wp + sChild[bb][d] * wc;
        }
    }
#pragma unroll
    for (int bb = 0; bb < 16; ++bb)
        pc[(size_t)(b0 + bb) * CP + p] = acc[bb];
}

// ---------------------------------------------------------------------------
// Kernel 3: fused GEMM chain. grid (4, B), block 256 (8 waves, 4x2 wave grid)
// Each block: rows [s0, s0+64) of batch b, all 512 columns.
// ---------------------------------------------------------------------------
#define KC 128            // K-chunk staged in LDS for GEMM1

__global__ __launch_bounds__(256)
void fused_mlp_kernel(const float*  __restrict__ x,
                      const __bf16* __restrict__ WhT,   // [P][D]
                      const __bf16* __restrict__ H0T,   // [P][P]
                      const __bf16* __restrict__ H1T,   // [P][P]
                      const float*  __restrict__ pc,    // [B][P]
                      const float*  __restrict__ scorer,// [P]
                      const unsigned char* __restrict__ mask, // [B][S]
                      float* __restrict__ escore)       // [B][SH]
{
    __shared__ __align__(16) __bf16 sA[64][KC + 8];     // staged x chunk (bf16)
    __shared__ __align__(16) __bf16 sC[64][CP + 8];     // composed (bf16)
    __shared__ float spc[CP];
    __shared__ float sScorer[CP];
    __shared__ float sScore[64];

    const int b    = blockIdx.y;
    const int s0   = blockIdx.x * 64;
    const int tid  = threadIdx.x;
    const int wave = tid >> 5;
    const int lane = tid & 31;
    const int ln   = lane & 15;
    const int hi   = lane >> 4;
    const int wm   = wave >> 1;   // 0..3 : 16-row tile
    const int wn   = wave & 1;    // 0..1 : 256-col half

    // Stage per-batch bias and scorer; zero score accumulator.
    for (int i = tid; i < CP; i += 256) {
        spc[i]     = pc[(size_t)b * CP + i];
        sScorer[i] = scorer[i];
    }
    if (tid < 64) sScore[tid] = 0.0f;

    const v8f vzero = {0.f, 0.f, 0.f, 0.f, 0.f, 0.f, 0.f, 0.f};
    v8f acc[16];
#pragma unroll
    for (int t = 0; t < 16; ++t) acc[t] = vzero;

    // ---------------- GEMM1: tanh(x @ Wh + pc) -> sC -----------------------
    for (int kc = 0; kc < CD; kc += KC) {
        __syncthreads();   // previous chunk fully consumed (and init visible)
        {
            const float* xb = x + ((size_t)(b * CS + s0)) * CD + kc;
#pragma unroll
            for (int it = 0; it < (64 * KC) / (256 * 4); ++it) {
                int idx = (tid + it * 256) * 4;
                int r = idx >> 7;          // / KC
                int c = idx & (KC - 1);
                float4 v = *(const float4*)(xb + (size_t)r * CD + c);
                v4bf w;
                w[0] = (__bf16)v.x; w[1] = (__bf16)v.y;
                w[2] = (__bf16)v.z; w[3] = (__bf16)v.w;
                *(v4bf*)&sA[r][c] = w;
            }
        }
        __syncthreads();
        for (int kk = 0; kk < KC; kk += 32) {
            Frag16 a;
            a.h[0] = *(const v8bf*)&sA[16 * wm + ln][kk + hi * 8];
            a.h[1] = *(const v8bf*)&sA[16 * wm + ln][kk + 16 + hi * 8];
#pragma unroll
            for (int t = 0; t < 16; ++t) {
                int n = wn * 256 + t * 16 + ln;
                const __bf16* wp = WhT + (size_t)n * CD + (kc + kk + hi * 8);
                Frag16 bb;
                bb.h[0] = *(const v8bf*)(wp);
                bb.h[1] = *(const v8bf*)(wp + 16);
                acc[t] = __builtin_amdgcn_wmma_f32_16x16x32_bf16(
                    false, a.v, false, bb.v, (short)0, acc[t], false, false);
            }
        }
    }
    // epilogue: bias + tanh -> bf16 in sC
#pragma unroll
    for (int t = 0; t < 16; ++t) {
        int n = wn * 256 + t * 16 + ln;
        float bias = spc[n];
#pragma unroll
        for (int i = 0; i < 8; ++i) {
            int r = 16 * wm + hi * 8 + i;
            sC[r][n] = (__bf16)fast_tanh(acc[t][i] + bias);
        }
    }
    __syncthreads();

    // ---------------- GEMM2: tanh(sC @ H0) -> sC (in place via regs) -------
#pragma unroll
    for (int t = 0; t < 16; ++t) acc[t] = vzero;
    for (int kk = 0; kk < CP; kk += 32) {
        Frag16 a;
        a.h[0] = *(const v8bf*)&sC[16 * wm + ln][kk + hi * 8];
        a.h[1] = *(const v8bf*)&sC[16 * wm + ln][kk + 16 + hi * 8];
#pragma unroll
        for (int t = 0; t < 16; ++t) {
            int n = wn * 256 + t * 16 + ln;
            const __bf16* wp = H0T + (size_t)n * CP + (kk + hi * 8);
            Frag16 bb;
            bb.h[0] = *(const v8bf*)(wp);
            bb.h[1] = *(const v8bf*)(wp + 16);
            acc[t] = __builtin_amdgcn_wmma_f32_16x16x32_bf16(
                false, a.v, false, bb.v, (short)0, acc[t], false, false);
        }
    }
    __syncthreads();   // all reads of sC complete before overwrite
#pragma unroll
    for (int t = 0; t < 16; ++t) {
        int n = wn * 256 + t * 16 + ln;
#pragma unroll
        for (int i = 0; i < 8; ++i)
            sC[16 * wm + hi * 8 + i][n] = (__bf16)fast_tanh(acc[t][i]);
    }
    __syncthreads();

    // ---------------- GEMM3: tanh(sC @ H1) then dot with scorer ------------
#pragma unroll
    for (int t = 0; t < 16; ++t) acc[t] = vzero;
    for (int kk = 0; kk < CP; kk += 32) {
        Frag16 a;
        a.h[0] = *(const v8bf*)&sC[16 * wm + ln][kk + hi * 8];
        a.h[1] = *(const v8bf*)&sC[16 * wm + ln][kk + 16 + hi * 8];
#pragma unroll
        for (int t = 0; t < 16; ++t) {
            int n = wn * 256 + t * 16 + ln;
            const __bf16* wp = H1T + (size_t)n * CP + (kk + hi * 8);
            Frag16 bb;
            bb.h[0] = *(const v8bf*)(wp);
            bb.h[1] = *(const v8bf*)(wp + 16);
            acc[t] = __builtin_amdgcn_wmma_f32_16x16x32_bf16(
                false, a.v, false, bb.v, (short)0, acc[t], false, false);
        }
    }
    // score partials: row r gets sum over this lane's columns
    {
        float part[8] = {0.f, 0.f, 0.f, 0.f, 0.f, 0.f, 0.f, 0.f};
#pragma unroll
        for (int t = 0; t < 16; ++t) {
            int n = wn * 256 + t * 16 + ln;
            float sv = sScorer[n];
#pragma unroll
            for (int i = 0; i < 8; ++i)
                part[i] += fast_tanh(acc[t][i]) * sv;
        }
#pragma unroll
        for (int i = 0; i < 8; ++i)
            atomicAdd(&sScore[16 * wm + hi * 8 + i], part[i]);
    }
    __syncthreads();

    // masked exp(score) -> workspace
    if (tid < 64) {
        int s = s0 + tid;
        if (s < CSH) {
            float sc = sScore[tid];
            float e  = mask[(size_t)b * CS + s] ? __expf(sc) : 0.0f;
            escore[(size_t)b * CSH + s] = e;
        }
    }
}

// ---------------------------------------------------------------------------
// Kernel 4: per-batch softmax normalization. grid B, block 256
// ---------------------------------------------------------------------------
__global__ __launch_bounds__(256)
void normalize_kernel(const float* __restrict__ escore,
                      float* __restrict__ out) {
    __shared__ float red[256];
    int b = blockIdx.x;
    int t = threadIdx.x;
    float v = (t < CSH) ? escore[(size_t)b * CSH + t] : 0.0f;
    red[t] = v;
    __syncthreads();
    for (int off = 128; off > 0; off >>= 1) {
        if (t < off) red[t] += red[t + off];
        __syncthreads();
    }
    float inv = 1.0f / (red[0] + EPSV);
    if (t < CSH) out[(size_t)b * CSH + t] = v * inv;
}

// ---------------------------------------------------------------------------
// Launch
// ---------------------------------------------------------------------------
extern "C" void kernel_launch(void* const* d_in, const int* in_sizes, int n_in,
                              void* d_out, int out_size, void* d_ws, size_t ws_size,
                              hipStream_t stream) {
    const float* x          = (const float*)d_in[0];
    const float* proj_head  = (const float*)d_in[1];
    const float* proj_prep  = (const float*)d_in[2];
    const float* proj_child = (const float*)d_in[3];
    const float* hidden     = (const float*)d_in[4];   // [2][P][P]
    const float* scorer     = (const float*)d_in[5];
    const unsigned char* mask = (const unsigned char*)d_in[6];
    float* out = (float*)d_out;

    // Workspace layout (bytes)
    char* ws = (char*)d_ws;
    __bf16* WhT = (__bf16*)(ws);                                  // P*D bf16 = 1 MiB
    __bf16* H0T = (__bf16*)(ws + (size_t)CP * CD * 2);            // P*P bf16
    __bf16* H1T = (__bf16*)(ws + (size_t)CP * CD * 2 + (size_t)CP * CP * 2);
    float*  pc  = (float*) (ws + (size_t)CP * CD * 2 + (size_t)CP * CP * 4);
    float*  esc = (float*) (ws + (size_t)CP * CD * 2 + (size_t)CP * CP * 4
                               + (size_t)CB * CP * 4);

    // 1) weight conversion/transposition to bf16
    {
        int n1 = CD * CP;
        transpose_to_bf16<<<(n1 + 255) / 256, 256, 0, stream>>>(proj_head, WhT, CD, CP);
        int n2 = CP * CP;
        transpose_to_bf16<<<(n2 + 255) / 256, 256, 0, stream>>>(hidden,            H0T, CP, CP);
        transpose_to_bf16<<<(n2 + 255) / 256, 256, 0, stream>>>(hidden + (size_t)CP * CP, H1T, CP, CP);
    }

    // 2) prep/child projection bias
    pc_kernel<<<dim3(2, CB / 16), 256, 0, stream>>>(x, proj_prep, proj_child, pc);

    // 3) fused GEMM chain -> masked exp scores
    fused_mlp_kernel<<<dim3(4, CB), 256, 0, stream>>>(
        x, WhT, H0T, H1T, pc, scorer, mask, esc);

    // 4) per-batch normalization
    normalize_kernel<<<CB, 256, 0, stream>>>(esc, out);
}
